// LightMetaPolicy_4612794876695
// MI455X (gfx1250) — matrix operand order repaired and checked by reading, not compiled, not run on hardware
//
#include <hip/hip_runtime.h>
#include <math.h>

// ---------------------------------------------------------------------------
// MI455X / gfx1250, wave32.  f16 WMMA (f32 accum) for all GEMMs.
// ---------------------------------------------------------------------------
typedef __attribute__((ext_vector_type(16))) _Float16 v16h;
typedef __attribute__((ext_vector_type(8)))  _Float16 v8h;
typedef __attribute__((ext_vector_type(4)))  _Float16 v4h;
typedef __attribute__((ext_vector_type(8)))  float    v8f;

#define TILE_E 32            // batch elements per workgroup
#define ROWS   160           // TILE_E * 5 agent rows
#define KDIM   128
#define PCOL   160           // fused projection cols: [q32 | k32 | v64 | rq32]
#define ASTR   136           // f16 LDS stride for activations (bank-conflict pad)
#define WSTR   136           // f16 LDS stride for WbigT
#define PSTR   164           // f32 LDS stride for projection output
#define HSTR   72            // f16 LDS stride for 64-wide tiles
#define HVSTR  68            // f32 LDS stride for hv

// ---- LDS carve (all 16B aligned) ----
#define OFF_WBIG  0                       // 160*136*2 = 43520
#define OFF_WCAT  43520                   // 128*72*2  = 18432
#define OFF_W2    61952                   // 32*72*2   = 4608
#define OFF_ABF   66560                   // 160*136*2 = 43520
#define OFF_PROJ  110080                  // 160*164*4 = 104960
#define OFF_PBF   215040                  // 32*72*2   = 4608
#define OFF_HBF   219648                  // 32*72*2   = 4608
#define OFF_HVF   224256                  // 32*68*4   = 8704
#define OFF_ASUM  232960                  // 160*4
#define OFF_LG    233600                  // 32*28*4
#define OFF_WB    237184                  // 32*8*4
#define OFF_VP    238208                  // 32*8*4
#define OFF_BBIG  239232                  // 160*4
#define OFF_BCAT  239872                  // 128*4
#define OFF_B2    240384                  // 32*4
#define OFF_WH2   240512                  // 64*4
#define OFF_BH2   240768                  // 16
#define SMEM_BYTES 240784
#define WCOPY_U4  4160                    // (43520+18432+4608)/16

__device__ __forceinline__ v8f wmma16(v16h a, v16h b, v8f c) {
  return __builtin_amdgcn_wmma_f32_16x16x32_f16(false, a, false, b, (short)0, c,
                                                false, false);
}

// A fragment (16x32 f16).  ISA layout: lane = h*16+m; elems 0..7 -> K=k0+h*8+e,
// elems 8..15 -> K=k0+16+h*8+(e-8).  Both halves contiguous -> 16B LDS loads.
__device__ __forceinline__ v16h frag_a(const _Float16* base, int stride,
                                       int row0, int k0, int lane) {
  int m = lane & 15, hh = lane >> 4;
  const _Float16* r = base + (row0 + m) * stride + k0 + hh * 8;
  v8h lo = *(const v8h*)(r);
  v8h hi = *(const v8h*)(r + 16);
  v16h a;
#pragma unroll
  for (int e = 0; e < 8; ++e) { a[e] = lo[e]; a[e + 8] = hi[e]; }
  return a;
}

// B fragment (32x16 f16), column-major weights.  lane = h*16+n holds column n,
// elems 0..7 -> K = k0 + h*8 + e, elems 8..15 -> K = k0 + 16 + h*8 + (e-8).
__device__ __forceinline__ v16h frag_b(const _Float16* base, int stride,
                                       int col0, int k0, int lane) {
  int n = lane & 15, hh = lane >> 4;
  const _Float16* r = base + (col0 + n) * stride + k0 + hh * 8;
  v8h lo = *(const v8h*)(r);
  v8h hi = *(const v8h*)(r + 16);
  v16h b;
#pragma unroll
  for (int e = 0; e < 8; ++e) { b[e] = lo[e]; b[e + 8] = hi[e]; }
  return b;
}

__device__ __forceinline__ float gelu_exact(float x) {
  return 0.5f * x * (1.0f + erff(x * 0.70710678118654752f));
}
__device__ __forceinline__ float sigmoidf(float x) {
  return 1.0f / (1.0f + __expf(-x));
}

// ---------------------------------------------------------------------------
// Prep kernel: build fused f16 weights in workspace.
//   WbigT [160 cols][136] : cols = [Wq | Wk | Wv | Wq@Wr]  (K contiguous)
//   bbig  [160]           : [bq | bk | bv | bq@Wr + br]
//   WcatT [128 cols][72]  : cols = [W1 | Wh1]
//   W2T   [32 cols][72]
// ---------------------------------------------------------------------------
__global__ void prep_kernel(
    const float* __restrict__ Wk, const float* __restrict__ bk,
    const float* __restrict__ Wq, const float* __restrict__ bq,
    const float* __restrict__ Wv, const float* __restrict__ bv,
    const float* __restrict__ Wr, const float* __restrict__ br,
    const float* __restrict__ W1, const float* __restrict__ b1,
    const float* __restrict__ W2, const float* __restrict__ b2,
    const float* __restrict__ Wh1, const float* __restrict__ bh1,
    const float* __restrict__ Wh2, const float* __restrict__ bh2,
    _Float16* __restrict__ WbigT, _Float16* __restrict__ WcatT,
    _Float16* __restrict__ W2T, float* __restrict__ bbig,
    float* __restrict__ bcat, float* __restrict__ b2d,
    float* __restrict__ wh2d, float* __restrict__ bh2d) {
  int tid = threadIdx.x;
  for (int idx = tid; idx < PCOL * KDIM; idx += 256) {
    int col = idx >> 7, k = idx & 127;
    float w;
    if (col < 32)       w = Wq[k * 32 + col];
    else if (col < 64)  w = Wk[k * 32 + (col - 32)];
    else if (col < 128) w = Wv[k * 64 + (col - 64)];
    else {
      int c = col - 128; float s = 0.f;
      for (int m = 0; m < 32; ++m) s += Wq[k * 32 + m] * Wr[m * 32 + c];
      w = s;
    }
    WbigT[col * WSTR + k] = (_Float16)w;
  }
  for (int idx = tid; idx < 128 * 64; idx += 256) {
    int col = idx >> 6, k = idx & 63;
    float w = (col < 64) ? W1[k * 64 + col] : Wh1[k * 64 + (col - 64)];
    WcatT[col * HSTR + k] = (_Float16)w;
  }
  for (int idx = tid; idx < 32 * 64; idx += 256) {
    int col = idx >> 6, k = idx & 63;
    W2T[col * HSTR + k] = (_Float16)W2[k * 32 + col];
  }
  if (tid < 160) {
    float b;
    if (tid < 32)       b = bq[tid];
    else if (tid < 64)  b = bk[tid - 32];
    else if (tid < 128) b = bv[tid - 64];
    else {
      int c = tid - 128; float s = br[c];
      for (int m = 0; m < 32; ++m) s += bq[m] * Wr[m * 32 + c];
      b = s;
    }
    bbig[tid] = b;
  }
  if (tid < 128) bcat[tid] = (tid < 64) ? b1[tid] : bh1[tid - 64];
  if (tid < 32)  b2d[tid] = b2[tid];
  if (tid < 64)  wh2d[tid] = Wh2[tid];
  if (tid == 0)  bh2d[0] = bh2[0];
}

// ---------------------------------------------------------------------------
// Main kernel: one workgroup = 32 batch elements (160 agent rows).
// ---------------------------------------------------------------------------
__global__ __launch_bounds__(256, 1)
void meta_policy_kernel(const float* __restrict__ x,
                        const _Float16* __restrict__ wAllG,  // WbigT|WcatT|W2T
                        const float* __restrict__ bbigG,
                        const float* __restrict__ bcatG,
                        const float* __restrict__ b2G,
                        const float* __restrict__ wh2G,
                        const float* __restrict__ bh2G,
                        float* __restrict__ out, int B) {
  extern __shared__ char smem[];
  _Float16* sWbig = (_Float16*)(smem + OFF_WBIG);
  _Float16* sWcat = (_Float16*)(smem + OFF_WCAT);
  _Float16* sW2   = (_Float16*)(smem + OFF_W2);
  _Float16* sAbf  = (_Float16*)(smem + OFF_ABF);
  float*    sProj = (float*)   (smem + OFF_PROJ);
  _Float16* sPbf  = (_Float16*)(smem + OFF_PBF);
  _Float16* sHbf  = (_Float16*)(smem + OFF_HBF);
  float*    sHv   = (float*)   (smem + OFF_HVF);
  float*    sAsum = (float*)   (smem + OFF_ASUM);
  float*    sLg   = (float*)   (smem + OFF_LG);
  float*    sWb   = (float*)   (smem + OFF_WB);
  float*    sVp   = (float*)   (smem + OFF_VP);
  float*    sBbig = (float*)   (smem + OFF_BBIG);
  float*    sBcat = (float*)   (smem + OFF_BCAT);
  float*    sB2   = (float*)   (smem + OFF_B2);
  float*    sWh2  = (float*)   (smem + OFF_WH2);
  float*    sBh2  = (float*)   (smem + OFF_BH2);

  const int tid  = threadIdx.x;
  const int lane = tid & 31;
  const int wave = tid >> 5;
  const int elemBase = blockIdx.x * TILE_E;
  if (elemBase >= B) return;

  // ---- Stage 0: zero mask accumulators, stage biases into LDS ----
  if (tid < 160) { sAsum[tid] = 0.f; sBbig[tid] = bbigG[tid]; }
  if (tid < 128) sBcat[tid] = bcatG[tid];
  if (tid < 32)  sB2[tid]   = b2G[tid];
  if (tid < 64)  sWh2[tid]  = wh2G[tid];
  if (tid == 0)  sBh2[0]    = bh2G[0];
  __syncthreads();

  // ---- Stage 1: weights -> LDS (single linear copy), x -> f16 LDS + |x| ----
  {
    const uint4* wsrc = (const uint4*)wAllG;
    uint4* wdst = (uint4*)(smem + OFF_WBIG);
    for (int i = tid; i < WCOPY_U4; i += 256) wdst[i] = wsrc[i];

    const float4* x4 = (const float4*)x;
    const size_t base4 = (size_t)elemBase * 160;  // 640 floats / elem = 160 f4
    for (int idx = tid; idx < ROWS * 32; idx += 256) {
      int r = idx >> 5, c4 = idx & 31;
      float4 v = x4[base4 + (size_t)r * 32 + c4];
      float s = fabsf(v.x) + fabsf(v.y) + fabsf(v.z) + fabsf(v.w);
      atomicAdd(&sAsum[r], s);             // ds_add_f32
      v4h h;
      h[0] = (_Float16)v.x; h[1] = (_Float16)v.y;
      h[2] = (_Float16)v.z; h[3] = (_Float16)v.w;
      *(v4h*)(sAbf + r * ASTR + c4 * 4) = h;
    }
  }
  __syncthreads();

  // ---- Stage 2: projection GEMM  [160,128] @ [128,160]  (WMMA f16) ----
  for (int p = wave; p < 100; p += 8) {
    int rt = p / 10, ct = p % 10;
    v8f acc = {};
#pragma unroll
    for (int ks = 0; ks < 4; ++ks) {
      v16h a = frag_a(sAbf, ASTR, rt * 16, ks * 32, lane);
      v16h b = frag_b(sWbig, WSTR, ct * 16, ks * 32, lane);
      acc = wmma16(a, b, acc);
    }
    int n = lane & 15, hh = lane >> 4;
    float bias = sBbig[ct * 16 + n];
#pragma unroll
    for (int r = 0; r < 8; ++r)
      sProj[(rt * 16 + r + hh * 8) * PSTR + ct * 16 + n] = acc[r] + bias;
  }
  __syncthreads();

  // ---- Stage 3: 5x5 attention + masked pool (8 threads / element) ----
  const int e = tid >> 3, t = tid & 7, rb = e * 5;
  for (int l = t; l < 25; l += 8) {                 // fused qk + rq.rq logits
    int i = l / 5, j = l % 5;
    const float* qi = sProj + (rb + i) * PSTR;       // q   @ col 0
    const float* kj = sProj + (rb + j) * PSTR + 32;  // k   @ col 32
    const float* ri = sProj + (rb + i) * PSTR + 128; // rq  @ col 128
    const float* rj = sProj + (rb + j) * PSTR + 128;
    float s = 0.f;
#pragma unroll 8
    for (int d = 0; d < 32; ++d) s += qi[d] * kj[d] + ri[d] * rj[d];
    sLg[e * 28 + l] = s * 0.17677669529663687f;      // 1/sqrt(32)
  }
  __syncthreads();
  if (t == 0) {                                      // softmax + mask collapse
    float mask[5], msum = 1e-8f;
#pragma unroll
    for (int a = 0; a < 5; ++a) {
      mask[a] = (sAsum[rb + a] > 0.1f) ? 1.f : 0.f;
      msum += mask[a];
    }
    float w[5] = {0.f, 0.f, 0.f, 0.f, 0.f};
#pragma unroll
    for (int i = 0; i < 5; ++i) {
      float row[5], mx = -1e30f;
#pragma unroll
      for (int j = 0; j < 5; ++j) { row[j] = sLg[e * 28 + i * 5 + j]; mx = fmaxf(mx, row[j]); }
      float es = 0.f;
#pragma unroll
      for (int j = 0; j < 5; ++j) { row[j] = __expf(row[j] - mx); es += row[j]; }
      float inv = mask[i] / es;
#pragma unroll
      for (int j = 0; j < 5; ++j) w[j] += row[j] * inv;
    }
#pragma unroll
    for (int j = 0; j < 5; ++j) sWb[e * 8 + j] = w[j] / msum;
  }
  __syncthreads();
  {                                                  // pooled = sum_j w_j * v_j
    float w0 = sWb[e * 8 + 0], w1 = sWb[e * 8 + 1], w2 = sWb[e * 8 + 2];
    float w3 = sWb[e * 8 + 3], w4 = sWb[e * 8 + 4];
#pragma unroll
    for (int q = 0; q < 8; ++q) {
      int d = t * 8 + q;
      float s = w0 * sProj[(rb + 0) * PSTR + 64 + d]
              + w1 * sProj[(rb + 1) * PSTR + 64 + d]
              + w2 * sProj[(rb + 2) * PSTR + 64 + d]
              + w3 * sProj[(rb + 3) * PSTR + 64 + d]
              + w4 * sProj[(rb + 4) * PSTR + 64 + d];
      sPbf[e * HSTR + d] = (_Float16)s;
    }
  }
  __syncthreads();

  // ---- Stage 4: [32,64] @ [64,128] = [h | hv], exact-erf GELU ----
  for (int p = wave; p < 16; p += 8) {
    int rt = p >> 3, ct = p & 7;
    v8f acc = {};
#pragma unroll
    for (int ks = 0; ks < 2; ++ks) {
      v16h a = frag_a(sPbf, HSTR, rt * 16, ks * 32, lane);
      v16h b = frag_b(sWcat, HSTR, ct * 16, ks * 32, lane);
      acc = wmma16(a, b, acc);
    }
    int n = lane & 15, hh = lane >> 4;
    int col = ct * 16 + n;
    float bias = sBcat[col];
#pragma unroll
    for (int r = 0; r < 8; ++r) {
      int row = rt * 16 + r + hh * 8;
      float g = gelu_exact(acc[r] + bias);
      if (col < 64) sHbf[row * HSTR + col] = (_Float16)g;
      else          sHv[row * HVSTR + (col - 64)] = g;
    }
  }
  __syncthreads();

  // ---- Stage 5: action head (WMMA, waves 0-3) + value head (all) ----
  for (int p = wave; p < 4; p += 8) {
    int rt = p >> 1, ct = p & 1;
    v8f acc = {};
#pragma unroll
    for (int ks = 0; ks < 2; ++ks) {
      v16h a = frag_a(sHbf, HSTR, rt * 16, ks * 32, lane);
      v16h b = frag_b(sW2, HSTR, ct * 16, ks * 32, lane);
      acc = wmma16(a, b, acc);
    }
    int n = lane & 15, hh = lane >> 4;
    int col = ct * 16 + n;
    float bias = sB2[col];
#pragma unroll
    for (int r = 0; r < 8; ++r) {
      int row = rt * 16 + r + hh * 8;
      out[(size_t)(elemBase + row) * 32 + col] = sigmoidf(acc[r] + bias);
    }
  }
  {
    float s = 0.f;
#pragma unroll
    for (int q = 0; q < 8; ++q) s += sHv[e * HVSTR + t * 8 + q] * sWh2[t * 8 + q];
    sVp[e * 8 + t] = s;
  }
  __syncthreads();
  if (t == 0) {
    float s = sBh2[0];
#pragma unroll
    for (int q = 0; q < 8; ++q) s += sVp[e * 8 + q];
    out[(size_t)B * 32 + (elemBase + e)] = s;
  }
}

// ---------------------------------------------------------------------------
extern "C" void kernel_launch(void* const* d_in, const int* in_sizes, int n_in,
                              void* d_out, int out_size, void* d_ws,
                              size_t ws_size, hipStream_t stream) {
  const float* x   = (const float*)d_in[0];
  const float* Wk  = (const float*)d_in[1];  const float* bk  = (const float*)d_in[2];
  const float* Wq  = (const float*)d_in[3];  const float* bq  = (const float*)d_in[4];
  const float* Wv  = (const float*)d_in[5];  const float* bv  = (const float*)d_in[6];
  const float* Wr  = (const float*)d_in[7];  const float* br  = (const float*)d_in[8];
  const float* W1  = (const float*)d_in[9];  const float* b1  = (const float*)d_in[10];
  const float* W2  = (const float*)d_in[11]; const float* b2  = (const float*)d_in[12];
  const float* Wh1 = (const float*)d_in[13]; const float* bh1 = (const float*)d_in[14];
  const float* Wh2 = (const float*)d_in[15]; const float* bh2 = (const float*)d_in[16];
  const int B = in_sizes[0] / 640;

  // workspace carve: [WbigT | WcatT | W2T] contiguous (single LDS copy), biases
  char* ws = (char*)d_ws;
  _Float16* WbigT = (_Float16*)(ws);
  _Float16* WcatT = (_Float16*)(ws + 43520);
  _Float16* W2T   = (_Float16*)(ws + 61952);
  float* bbig = (float*)(ws + 66560);
  float* bcat = (float*)(ws + 67200);
  float* b2d  = (float*)(ws + 67712);
  float* wh2d = (float*)(ws + 67840);
  float* bh2d = (float*)(ws + 68096);

  prep_kernel<<<1, 256, 0, stream>>>(Wk, bk, Wq, bq, Wv, bv, Wr, br, W1, b1,
                                     W2, b2, Wh1, bh1, Wh2, bh2, WbigT, WcatT,
                                     W2T, bbig, bcat, b2d, wh2d, bh2d);

  (void)hipFuncSetAttribute((const void*)meta_policy_kernel,
                            hipFuncAttributeMaxDynamicSharedMemorySize,
                            SMEM_BYTES);
  int nblk = (B + TILE_E - 1) / TILE_E;
  meta_policy_kernel<<<nblk, 256, SMEM_BYTES, stream>>>(
      x, WbigT, bbig, bcat, b2d, wh2d, bh2d, (float*)d_out, B);
}
